// AttentionFusion_20452634263891
// MI455X (gfx1250) — matrix operand order repaired and checked by reading, not compiled, not run on hardware
//
#include <hip/hip_runtime.h>
#include <hip/hip_bf16.h>
#include <math.h>

typedef __bf16 bf16;
typedef __attribute__((ext_vector_type(16))) __bf16 v16bf;
typedef __attribute__((ext_vector_type(8)))  __bf16 v8bf;
typedef __attribute__((ext_vector_type(4)))  __bf16 v4bf;
typedef __attribute__((ext_vector_type(8)))  float  v8f;

union BFrag16 { v16bf v; v8bf h[2]; };

static constexpr int HIDDEN = 1024;
static constexpr int NUM_HEADS = 16;
static constexpr int HEAD_DIM = 64;
static constexpr int SEQ = 2048;
static constexpr int BATCH = 2;
static constexpr int ROWS = BATCH * SEQ;       // 4096
static constexpr int QKV_N = 3 * HIDDEN;       // 3072

// ---------------------------------------------------------------------------
// fp32 -> bf16 conversion (vectorized x4)
// ---------------------------------------------------------------------------
__global__ __launch_bounds__(256) void cvt_f32_bf16(const float* __restrict__ in,
                                                    bf16* __restrict__ out, int n4) {
  int i = blockIdx.x * 256 + threadIdx.x;
  if (i < n4) {
    float4 f = ((const float4*)in)[i];
    v4bf o;
    o[0] = (bf16)f.x; o[1] = (bf16)f.y; o[2] = (bf16)f.z; o[3] = (bf16)f.w;
    ((v4bf*)out)[i] = o;
  }
}

// ---------------------------------------------------------------------------
// Generic C[M,N] = A[M,K] * B[N,K]^T + bias, bf16 inputs, fp32 accumulate.
// One wave per 64x64 C tile (4x4 WMMA accumulators); 16 WMMA per k-step of 32.
// M,N multiples of 64; K multiple of 32.
// ---------------------------------------------------------------------------
template <bool OUT_F32>
__global__ __launch_bounds__(256) void gemm_bf16_nt(
    const bf16* __restrict__ A, const bf16* __restrict__ Bm,
    const float* __restrict__ bias, void* __restrict__ Cout,
    int M, int N, int K) {
  const int lane = threadIdx.x & 31;
  const int wave = threadIdx.x >> 5;
  const int r    = lane & 15;
  const int half = lane >> 4;

  const int tilesN = N >> 6;
  const int w  = blockIdx.x * 8 + wave;
  const int tm = (w / tilesN) * 64;
  const int tn = (w % tilesN) * 64;

  v8f acc[4][4] = {};

  // A-fragment base: row tm+r, 16 contiguous K at half*16 (matches B layout)
  const bf16* pa = A + (size_t)(tm + r) * K + half * 16;
  // B-fragment base: column tn+r, 16 contiguous K at half*16 (same contiguous k order as A)
  const bf16* pb = Bm + (size_t)(tn + r) * K + half * 16;

  for (int k0 = 0; k0 < K; k0 += 32) {
    BFrag16 a[4], b[4];
    #pragma unroll
    for (int i = 0; i < 4; i++) {
      const bf16* p = pa + (size_t)i * 16 * K + k0;
      a[i].h[0] = *(const v8bf*)(p);
      a[i].h[1] = *(const v8bf*)(p + 8);
    }
    #pragma unroll
    for (int j = 0; j < 4; j++) {
      b[j].v = *(const v16bf*)(pb + (size_t)j * 16 * K + k0);
    }
    __builtin_prefetch(pa + k0 + 64, 0, 1);
    __builtin_prefetch(pb + k0 + 64, 0, 1);
    #pragma unroll
    for (int i = 0; i < 4; i++) {
      #pragma unroll
      for (int j = 0; j < 4; j++) {
        acc[i][j] = __builtin_amdgcn_wmma_f32_16x16x32_bf16(
            false, a[i].v, false, b[j].v, (short)0, acc[i][j], false, false);
      }
    }
  }

  // C layout: VGPR t <-> row t + 8*half; column = lane&15
  #pragma unroll
  for (int i = 0; i < 4; i++) {
    #pragma unroll
    for (int j = 0; j < 4; j++) {
      const int col = tn + j * 16 + r;
      const float bv = bias[col];
      const int row0 = tm + i * 16 + half * 8;
      #pragma unroll
      for (int t = 0; t < 8; t++) {
        float v = acc[i][j][t] + bv;
        if (OUT_F32)
          ((float*)Cout)[(size_t)(row0 + t) * N + col] = v;
        else
          ((bf16*)Cout)[(size_t)(row0 + t) * N + col] = (bf16)v;
      }
    }
  }
}

// ---------------------------------------------------------------------------
// Flash attention: one wave per 16-query tile, 8 waves/block share (b,h).
// Streams 64-key blocks; K tile and transposed V tile staged in LDS;
// P re-laid-out (C-layout -> A-layout) via per-wave LDS tiles.
// ---------------------------------------------------------------------------
__global__ __launch_bounds__(256) void attention_kernel(
    const bf16* __restrict__ qkv, bf16* __restrict__ O) {
  __shared__ bf16 Kt[64][80];         // [key][d]   (pad rows to 160B, 32B-aligned)
  __shared__ bf16 Vt[64][80];         // [d][key]   transposed V tile
  __shared__ bf16 Pbuf[8][16 * 64];   // per-wave P tile, row-major [q][key]

  const int lane = threadIdx.x & 31;
  const int wave = threadIdx.x >> 5;
  const int r    = lane & 15;
  const int half = lane >> 4;

  const int bh   = blockIdx.x >> 4;   // 0..31  (b*16 + h)
  const int qblk = blockIdx.x & 15;
  const int b    = bh >> 4;
  const int h    = bh & 15;
  const int q0   = qblk * 128 + wave * 16;

  const size_t rs = QKV_N;  // row stride in qkv buffer
  const bf16* Qb = qkv + (size_t)(b * SEQ) * rs + h * HEAD_DIM;
  const bf16* Kb = Qb + HIDDEN;
  const bf16* Vb = Qb + 2 * HIDDEN;

  // Q fragments (A layout), held in registers for the whole kernel
  BFrag16 qa[2];
  {
    const bf16* pq = Qb + (size_t)(q0 + r) * rs + half * 16;
    qa[0].h[0] = *(const v8bf*)(pq);
    qa[0].h[1] = *(const v8bf*)(pq + 8);
    qa[1].h[0] = *(const v8bf*)(pq + 32);
    qa[1].h[1] = *(const v8bf*)(pq + 40);
  }

  v8f o[4] = {};
  float mrun[8], lrun[8];
  #pragma unroll
  for (int t = 0; t < 8; t++) { mrun[t] = -3.0e38f; lrun[t] = 0.0f; }
  const float scale = 0.125f;  // 1/sqrt(64)

  bf16* Pw = &Pbuf[wave][0];

  for (int kb = 0; kb < SEQ; kb += 64) {
    // --- cooperative staging -------------------------------------------------
    // K tile: 64x64 bf16, row-major; one 32B load + one 32B LDS store per thread
    {
      const int row = threadIdx.x >> 2;         // 0..63
      const int c   = (threadIdx.x & 3) * 16;   // 0,16,32,48
      BFrag16 tmp;
      tmp.v = *(const v16bf*)(Kb + (size_t)(kb + row) * rs + c);
      *(v16bf*)(&Kt[row][c]) = tmp.v;
    }
    // V tile transposed (Vt[d][k] = V[kb+k][d]), pair-packed b32 LDS stores
    {
      const int kp = threadIdx.x >> 3;          // 0..31 -> keys 2kp, 2kp+1
      const int d0 = (threadIdx.x & 7) * 8;     // 0..56
      v8bf v0 = *(const v8bf*)(Vb + (size_t)(kb + 2 * kp) * rs + d0);
      v8bf v1 = *(const v8bf*)(Vb + (size_t)(kb + 2 * kp + 1) * rs + d0);
      #pragma unroll
      for (int j = 0; j < 8; j++) {
        unsigned u = (unsigned)__builtin_bit_cast(unsigned short, v0[j]) |
                     ((unsigned)__builtin_bit_cast(unsigned short, v1[j]) << 16);
        *(unsigned*)(&Vt[d0 + j][2 * kp]) = u;
      }
    }
    __syncthreads();

    // --- scores S = Q K^T, four 16-key column tiles, B-frags from LDS --------
    v8f s[4] = {};
    #pragma unroll
    for (int kc = 0; kc < 4; kc++) {
      const bf16* pk = &Kt[kc * 16 + r][half * 16];
      BFrag16 b0, b1;
      b0.v = *(const v16bf*)(pk);        // contraction d = 0..31
      b1.v = *(const v16bf*)(pk + 32);   // contraction d = 32..63
      s[kc] = __builtin_amdgcn_wmma_f32_16x16x32_bf16(false, qa[0].v, false, b0.v,
                                                      (short)0, s[kc], false, false);
      s[kc] = __builtin_amdgcn_wmma_f32_16x16x32_bf16(false, qa[1].v, false, b1.v,
                                                      (short)0, s[kc], false, false);
    }

    // --- online softmax; row t+8*half lives in one 16-lane half --------------
    #pragma unroll
    for (int t = 0; t < 8; t++) {
      float sv0 = s[0][t] * scale;
      float sv1 = s[1][t] * scale;
      float sv2 = s[2][t] * scale;
      float sv3 = s[3][t] * scale;
      float mx = fmaxf(fmaxf(sv0, sv1), fmaxf(sv2, sv3));
      #pragma unroll
      for (int off = 8; off >= 1; off >>= 1) mx = fmaxf(mx, __shfl_xor(mx, off, 32));
      const float mnew = fmaxf(mrun[t], mx);
      const float corr = __expf(mrun[t] - mnew);
      const float p0 = __expf(sv0 - mnew);
      const float p1 = __expf(sv1 - mnew);
      const float p2 = __expf(sv2 - mnew);
      const float p3 = __expf(sv3 - mnew);
      float rsum = (p0 + p1) + (p2 + p3);
      #pragma unroll
      for (int off = 8; off >= 1; off >>= 1) rsum += __shfl_xor(rsum, off, 32);
      lrun[t] = lrun[t] * corr + rsum;
      mrun[t] = mnew;
      #pragma unroll
      for (int dt = 0; dt < 4; dt++) o[dt][t] *= corr;
      const int prow = t + half * 8;
      Pw[prow * 64 + r]      = (bf16)p0;
      Pw[prow * 64 + 16 + r] = (bf16)p1;
      Pw[prow * 64 + 32 + r] = (bf16)p2;
      Pw[prow * 64 + 48 + r] = (bf16)p3;
    }

    // --- O += P @ V : A = P tile from LDS, B = Vt rows (contiguous keys) -----
    BFrag16 pa0, pa1;
    {
      const bf16* pp = Pw + r * 64 + half * 16;
      pa0.h[0] = *(const v8bf*)(pp);          // keys 0..31
      pa0.h[1] = *(const v8bf*)(pp + 8);
      pa1.h[0] = *(const v8bf*)(pp + 32);     // keys 32..63
      pa1.h[1] = *(const v8bf*)(pp + 40);
    }
    #pragma unroll
    for (int dt = 0; dt < 4; dt++) {
      BFrag16 bv0, bv1;
      bv0.v = *(const v16bf*)(&Vt[dt * 16 + r][half * 16]);        // keys 0..31
      bv1.v = *(const v16bf*)(&Vt[dt * 16 + r][32 + half * 16]);   // keys 32..63
      o[dt] = __builtin_amdgcn_wmma_f32_16x16x32_bf16(false, pa0.v, false, bv0.v,
                                                      (short)0, o[dt], false, false);
      o[dt] = __builtin_amdgcn_wmma_f32_16x16x32_bf16(false, pa1.v, false, bv1.v,
                                                      (short)0, o[dt], false, false);
    }
    __syncthreads();
  }

  // normalize and store O (bf16) into [4096, 1024] at head offset
  #pragma unroll
  for (int t = 0; t < 8; t++) {
    const float inv = 1.0f / lrun[t];
    const int row = q0 + t + half * 8;
    const size_t base = (size_t)(b * SEQ + row) * HIDDEN + h * HEAD_DIM;
    #pragma unroll
    for (int dt = 0; dt < 4; dt++) {
      O[base + dt * 16 + r] = (bf16)(o[dt][t] * inv);
    }
  }
}

// ---------------------------------------------------------------------------
extern "C" void kernel_launch(void* const* d_in, const int* in_sizes, int n_in,
                              void* d_out, int out_size, void* d_ws, size_t ws_size,
                              hipStream_t stream) {
  const float* x     = (const float*)d_in[0];
  const float* qkv_w = (const float*)d_in[1];
  const float* qkv_b = (const float*)d_in[2];
  const float* out_w = (const float*)d_in[3];
  const float* out_b = (const float*)d_in[4];

  char* ws = (char*)d_ws;
  size_t off = 0;
  bf16* xb    = (bf16*)(ws + off); off += (size_t)ROWS * HIDDEN * 2;    // 8 MB
  bf16* wqkvb = (bf16*)(ws + off); off += (size_t)QKV_N * HIDDEN * 2;   // 6 MB
  bf16* woutb = (bf16*)(ws + off); off += (size_t)HIDDEN * HIDDEN * 2;  // 2 MB
  bf16* qkv   = (bf16*)(ws + off); off += (size_t)ROWS * QKV_N * 2;     // 24 MB
  bf16* attn  = (bf16*)(ws + off);                                      // 8 MB

  // fp32 -> bf16 conversions
  cvt_f32_bf16<<<(ROWS * HIDDEN / 4 + 255) / 256, 256, 0, stream>>>(x, xb, ROWS * HIDDEN / 4);
  cvt_f32_bf16<<<(QKV_N * HIDDEN / 4 + 255) / 256, 256, 0, stream>>>(qkv_w, wqkvb, QKV_N * HIDDEN / 4);
  cvt_f32_bf16<<<(HIDDEN * HIDDEN / 4 + 255) / 256, 256, 0, stream>>>(out_w, woutb, HIDDEN * HIDDEN / 4);

  // QKV projection: [4096,1024] x [3072,1024]^T + bias -> bf16 [4096,3072]
  {
    int blocks = (ROWS / 64) * (QKV_N / 64) / 8;  // 384
    gemm_bf16_nt<false><<<blocks, 256, 0, stream>>>(xb, wqkvb, qkv_b, qkv,
                                                    ROWS, QKV_N, HIDDEN);
  }

  // fused attention -> bf16 [4096,1024]
  attention_kernel<<<BATCH * NUM_HEADS * (SEQ / 128), 256, 0, stream>>>(qkv, attn);

  // output projection: [4096,1024] x [1024,1024]^T + bias -> fp32 d_out
  {
    int blocks = (ROWS / 64) * (HIDDEN / 64) / 8;  // 128
    gemm_bf16_nt<true><<<blocks, 256, 0, stream>>>(attn, woutb, out_b, d_out,
                                                   ROWS, HIDDEN, HIDDEN);
  }
}
